// Hamiltonian_49340584296467
// MI455X (gfx1250) — matrix operand order repaired and checked by reading, not compiled, not run on hardware
//
#include <hip/hip_runtime.h>

typedef __attribute__((ext_vector_type(16))) _Float16 v16h;
typedef __attribute__((ext_vector_type(8)))  _Float16 v8h;
typedef __attribute__((ext_vector_type(8)))  float    v8f;

#define KDIM 100           // hidden width
#define NT   7             // N tiles of 16 -> 112 padded
#define NPAD (NT * 16)     // 112
#define HSTR 136           // f16 per staging row (128 + 8 pad -> bank-conflict free)
#define WAVES 8
#define TILES_PER_WAVE 4   // iterations; each iteration = 32 rows (2 M-tiles)
#define ROWS_PER_WG (WAVES * 32 * TILES_PER_WAVE)   // 1024

union AF { v16h v; v8h h[2]; };

__device__ __forceinline__ v8f vzero8() {
  v8f z = {0.f, 0.f, 0.f, 0.f, 0.f, 0.f, 0.f, 0.f};
  return z;
}

// Hardware tanh (V_TANH_F32, CDNA5 TRANS op) with branch-free fallback.
__device__ __forceinline__ float fast_tanh(float x) {
#if __has_builtin(__builtin_amdgcn_tanhf)
  return __builtin_amdgcn_tanhf(x);
#elif __has_builtin(__builtin_amdgcn_tanh_f32)
  return __builtin_amdgcn_tanh_f32(x);
#else
  const float e = __builtin_amdgcn_exp2f(x * 2.8853900817779268f);
  const float t = __builtin_amdgcn_rcpf(e + 1.0f);
  return __builtin_fmaf(-2.0f, t, 1.0f);
#endif
}

__global__ __launch_bounds__(256)
void ham_fused_wmma_kernel(const float* __restrict__ Xg,
                           const float* __restrict__ Gm,
                           const float* __restrict__ Bv,
                           const float* __restrict__ W1, const float* __restrict__ b1,
                           const float* __restrict__ W2, const float* __restrict__ b2,
                           const float* __restrict__ W3, const float* __restrict__ b3,
                           const float* __restrict__ W4,
                           float* __restrict__ outp)
{
  // ---- LDS: weights pre-swizzled into WMMA B-fragment order + per-wave staging ----
  __shared__ __align__(32) _Float16 sW1[7  * 512];   //  7 KiB  (kc=0,    nt=0..6)
  __shared__ __align__(32) _Float16 sW2[28 * 512];   // 28 KiB  (kc=0..3, nt=0..6)
  __shared__ __align__(32) _Float16 sW3[28 * 512];   // 28 KiB
  __shared__ __align__(32) _Float16 sW4f[4 * 512];   //  4 KiB  (kc=0..3, nt=0, N-pad 1->16)
  __shared__ float sB1[NPAD], sB2[NPAD], sB3[NPAD];
  __shared__ __align__(32) _Float16 sH[WAVES][32 * HSTR];  // per-wave staging (2 M-tiles)
  __shared__ float sKV[WAVES][32];                         // layer-4 result bounce

  const int tid  = threadIdx.x;
  const int wave = tid >> 5;
  const int lane = tid & 31;

  // B-fragment swizzle fill: block (kc,nt) is a 32(K)x16(N) fragment; lane l<16 owns
  // N=l, K = kc*32 + {0..7,16..23} ; lane l>=16 owns N=l-16, K = kc*32 + {8..15,24..31}.
  auto fillW = [&](const float* W, _Float16* dst, int Kd, int Nd, int nkc, int nnt) {
    for (int e = tid; e < nkc * nnt * 512; e += 256) {
      const int blk = e >> 9;
      const int kc  = blk / nnt;
      const int nt  = blk % nnt;
      const int ln  = (e >> 4) & 31;
      const int i   = e & 15;
      const int k   = kc * 32 + (i & 7) + ((ln >> 4) << 3) + ((i >> 3) << 4);
      const int n   = nt * 16 + (ln & 15);
      const float v = (k < Kd && n < Nd) ? W[n * Kd + k] : 0.f;
      dst[e] = (_Float16)v;
    }
  };
  fillW(W1, sW1,  6,    KDIM, 1, NT);
  fillW(W2, sW2,  KDIM, KDIM, 4, NT);
  fillW(W3, sW3,  KDIM, KDIM, 4, NT);
  fillW(W4, sW4f, KDIM, 1,    4, 1);     // W4 is (1,100): n==0 column only
  for (int e = tid; e < NPAD; e += 256) {
    sB1[e] = (e < KDIM) ? b1[e] : 0.f;
    sB2[e] = (e < KDIM) ? b2[e] : 0.f;
    sB3[e] = (e < KDIM) ? b3[e] : 0.f;
  }
  { // zero this wave's staging (incl. K-pad cols 100..127 read by kc=3)
    unsigned* hz = (unsigned*)&sH[wave][0];
    for (int i = lane; i < (32 * HSTR) / 2; i += 32) hz[i] = 0u;
  }
  __syncthreads();

  // constant 2x2 M = Gamma^T Gamma + diag(B)^2
  const float g00 = Gm[0], g01 = Gm[1], g10 = Gm[2], g11 = Gm[3];
  const float Bd0 = Bv[0], Bd1 = Bv[1];
  const float M00 = g00 * g00 + g10 * g10 + Bd0 * Bd0;
  const float M01 = g00 * g01 + g10 * g11;          // == M10
  const float M11 = g01 * g01 + g11 * g11 + Bd1 * Bd1;

  const long wgBase = (long)blockIdx.x * ROWS_PER_WG;
  _Float16* H = &sH[wave][0];

  for (int it = 0; it < TILES_PER_WAVE; ++it) {
    const long tileBase = wgBase + (long)(it * WAVES + wave) * 32;

    if (it + 1 < TILES_PER_WAVE)                // prefetch next X super-tile
      __builtin_prefetch(Xg + (tileBase + (long)WAVES * 32 + lane) * 12, 0, 0);

    // ============ stage A: per-row Schur-complement solve (all 32 lanes) =============
    float kin;
    {
      const long row = tileBase + lane;
      const float4* Xv = (const float4*)(Xg + row * 12);
      const float4 a0 = Xv[0], a1 = Xv[1], a2 = Xv[2];
      const float q0x = a0.x, q0y = a0.y, q0z = a0.z;
      const float p0x = a0.w, p0y = a1.x, p0z = a1.y;
      const float q1x = a1.z, q1y = a1.w, q1z = a2.x;
      const float p1x = a2.y, p1y = a2.z, p1z = a2.w;

      // MM = [[M00*I, M01*(I-q0q0^T)], [M01*(I-q1q1^T), M11*I]] ; solve MM x = p.
      // Schur: S = M11*I - c*(E1*E0), c = M01^2/M00,
      //        E1*E0 = I - q0q0^T - q1q1^T + (q0.q1) q1 q0^T
      const float c   = M01 * M01 / M00;
      const float d01 = q0x * q1x + q0y * q1y + q0z * q1z;
      const float S00 = M11 - c * (1.f - q0x*q0x - q1x*q1x + d01*q1x*q0x);
      const float S01 =       -c * (     - q0x*q0y - q1x*q1y + d01*q1x*q0y);
      const float S02 =       -c * (     - q0x*q0z - q1x*q1z + d01*q1x*q0z);
      const float S10 =       -c * (     - q0y*q0x - q1y*q1x + d01*q1y*q0x);
      const float S11 = M11 - c * (1.f - q0y*q0y - q1y*q1y + d01*q1y*q0y);
      const float S12 =       -c * (     - q0y*q0z - q1y*q1z + d01*q1y*q0z);
      const float S20 =       -c * (     - q0z*q0x - q1z*q1x + d01*q1z*q0x);
      const float S21 =       -c * (     - q0z*q0y - q1z*q1y + d01*q1z*q0y);
      const float S22 = M11 - c * (1.f - q0z*q0z - q1z*q1z + d01*q1z*q0z);

      // rhs = p1 - (M01/M00) * (E1 p0),  E1 v = v - q1 (q1.v)
      const float r  = M01 / M00;
      const float qp = q1x * p0x + q1y * p0y + q1z * p0z;
      const float r0 = p1x - r * (p0x - q1x * qp);
      const float r1 = p1y - r * (p0y - q1y * qp);
      const float r2 = p1z - r * (p0z - q1z * qp);

      // Cramer 3x3
      const float c00 = S11 * S22 - S12 * S21;
      const float c01 = S10 * S22 - S12 * S20;
      const float c02 = S10 * S21 - S11 * S20;
      const float det = S00 * c00 - S01 * c01 + S02 * c02;
      const float inv = 1.f / det;
      const float x20 = (r0 * c00 - S01 * (r1 * S22 - S12 * r2) + S02 * (r1 * S21 - S11 * r2)) * inv;
      const float x21 = (S00 * (r1 * S22 - S12 * r2) - r0 * c01 + S02 * (S10 * r2 - r1 * S20)) * inv;
      const float x22 = (S00 * (S11 * r2 - r1 * S21) - S01 * (S10 * r2 - r1 * S20) + r0 * c02) * inv;

      // x1 = (p0 - M01 * E0 x2)/M00 ;  E0 v = v - q0 (q0.v)
      const float qx = q0x * x20 + q0y * x21 + q0z * x22;
      const float x10 = (p0x - M01 * (x20 - q0x * qx)) / M00;
      const float x11 = (p0y - M01 * (x21 - q0y * qx)) / M00;
      const float x12 = (p0z - M01 * (x22 - q0z * qx)) / M00;

      kin = 0.5f * (p0x*x10 + p0y*x11 + p0z*x12 + p1x*x20 + p1y*x21 + p1z*x22);

      // stage qf (6 vals) as f16 into this lane's A row; re-zero K-pad cols 6..31
      _Float16* hr = H + lane * HSTR;
      hr[0] = (_Float16)q0x; hr[1] = (_Float16)q0y; hr[2] = (_Float16)q0z;
      hr[3] = (_Float16)q1x; hr[4] = (_Float16)q1y; hr[5] = (_Float16)q1z;
      unsigned* hz = (unsigned*)(hr + 6);   // (row*136+6)*2 is 4B aligned
      #pragma unroll
      for (int z = 0; z < 13; ++z) hz[z] = 0u;
    }

    // ============ MLP: two 16-row M-tiles share every B fragment ====================
    v8f acc0[NT], acc1[NT];
    const int ar = lane & 15;
    const int kb = (lane < 16) ? 0 : 8;     // A layout: K {0..7,16..23} / {8..15,24..31}
    const int mb = (lane < 16) ? 0 : 8;     // C layout: VGPR r -> row mb+r, col = lane&15
    _Float16* H1 = H + 16 * HSTR;           // second M-tile rows (written by act_store)

    // layer 1: K-chunk 0 only (qf padded 6->32)
    {
      asm volatile("s_wait_dscnt 0" ::: "memory");
      const _Float16* ap0 = H  + ar * HSTR + kb;
      const _Float16* ap1 = H1 + ar * HSTR + kb;
      AF a0; a0.h[0] = *(const v8h*)ap0; a0.h[1] = *(const v8h*)(ap0 + 16);
      AF a1; a1.h[0] = *(const v8h*)ap1; a1.h[1] = *(const v8h*)(ap1 + 16);
      #pragma unroll
      for (int nt = 0; nt < NT; ++nt) {
        const v16h b = *(const v16h*)(sW1 + (nt * 512 + lane * 16));
        acc0[nt] = __builtin_amdgcn_wmma_f32_16x16x32_f16(false, a0.v, false, b,
                                                          (short)0, vzero8(), false, false);
        acc1[nt] = __builtin_amdgcn_wmma_f32_16x16x32_f16(false, a1.v, false, b,
                                                          (short)0, vzero8(), false, false);
      }
    }

    auto act_store = [&](const float* bias) {  // bias + hw tanh, restage both tiles
      const int nc0 = lane & 15;
      #pragma unroll
      for (int nt = 0; nt < NT; ++nt) {
        const float bb = bias[nt * 16 + nc0];
        #pragma unroll
        for (int rr = 0; rr < 8; ++rr) {
          H [(mb + rr) * HSTR + nt * 16 + nc0] = (_Float16)fast_tanh(acc0[nt][rr] + bb);
          H1[(mb + rr) * HSTR + nt * 16 + nc0] = (_Float16)fast_tanh(acc1[nt][rr] + bb);
        }
      }
    };

    auto gemm_layer = [&](const _Float16* Wl) {  // 2x (16x112 += 16x128 * 128x112)
      #pragma unroll
      for (int nt = 0; nt < NT; ++nt) { acc0[nt] = vzero8(); acc1[nt] = vzero8(); }
      asm volatile("s_wait_dscnt 0" ::: "memory");
      #pragma unroll
      for (int kc = 0; kc < 4; ++kc) {
        const _Float16* ap0 = H  + ar * HSTR + kc * 32 + kb;
        const _Float16* ap1 = H1 + ar * HSTR + kc * 32 + kb;
        AF a0; a0.h[0] = *(const v8h*)ap0; a0.h[1] = *(const v8h*)(ap0 + 16);
        AF a1; a1.h[0] = *(const v8h*)ap1; a1.h[1] = *(const v8h*)(ap1 + 16);
        #pragma unroll
        for (int nt = 0; nt < NT; ++nt) {
          const v16h b = *(const v16h*)(Wl + ((kc * NT + nt) * 512 + lane * 16));
          acc0[nt] = __builtin_amdgcn_wmma_f32_16x16x32_f16(false, a0.v, false, b,
                                                            (short)0, acc0[nt], false, false);
          acc1[nt] = __builtin_amdgcn_wmma_f32_16x16x32_f16(false, a1.v, false, b,
                                                            (short)0, acc1[nt], false, false);
        }
      }
    };

    act_store(sB1);
    gemm_layer(sW2);
    act_store(sB2);
    gemm_layer(sW3);
    act_store(sB3);

    // ============ layer 4 as WMMA: v = h3 . W4 (N padded 1->16) ======================
    {
      v8f a40 = vzero8(), a41 = vzero8();
      asm volatile("s_wait_dscnt 0" ::: "memory");
      #pragma unroll
      for (int kc = 0; kc < 4; ++kc) {
        const _Float16* ap0 = H  + ar * HSTR + kc * 32 + kb;
        const _Float16* ap1 = H1 + ar * HSTR + kc * 32 + kb;
        AF a0; a0.h[0] = *(const v8h*)ap0; a0.h[1] = *(const v8h*)(ap0 + 16);
        AF a1; a1.h[0] = *(const v8h*)ap1; a1.h[1] = *(const v8h*)(ap1 + 16);
        const v16h b = *(const v16h*)(sW4f + (kc * 512 + lane * 16));
        a40 = __builtin_amdgcn_wmma_f32_16x16x32_f16(false, a0.v, false, b,
                                                     (short)0, a40, false, false);
        a41 = __builtin_amdgcn_wmma_f32_16x16x32_f16(false, a1.v, false, b,
                                                     (short)0, a41, false, false);
      }
      // column N==0 holds the result: lanes 0 (rows 0..7) and 16 (rows 8..15)
      if ((lane & 15) == 0) {
        #pragma unroll
        for (int rr = 0; rr < 8; ++rr) {
          sKV[wave][mb + rr]      = a40[rr];
          sKV[wave][16 + mb + rr] = a41[rr];
        }
      }
      asm volatile("s_wait_dscnt 0" ::: "memory");
      outp[tileBase + lane] = sKV[wave][lane] + kin;
    }
  }
}

extern "C" void kernel_launch(void* const* d_in, const int* in_sizes, int n_in,
                              void* d_out, int out_size, void* d_ws, size_t ws_size,
                              hipStream_t stream) {
  const float* X   = (const float*)d_in[0];
  const float* Gm  = (const float*)d_in[1];
  const float* Bv  = (const float*)d_in[2];
  const float* W1  = (const float*)d_in[3];
  const float* b1  = (const float*)d_in[4];
  const float* W2  = (const float*)d_in[5];
  const float* b2  = (const float*)d_in[6];
  const float* W3  = (const float*)d_in[7];
  const float* b3  = (const float*)d_in[8];
  const float* W4  = (const float*)d_in[9];

  const int batch = in_sizes[0] / 12;              // 1048576
  const int grid  = batch / ROWS_PER_WG;           // 1024 workgroups

  ham_fused_wmma_kernel<<<grid, 256, 0, stream>>>(X, Gm, Bv, W1, b1, W2, b2, W3, b3, W4,
                                                  (float*)d_out);
}
